// HeteroGNN_10900626997402
// MI455X (gfx1250) — compile-verified, matrix operand order checked by reading, not run on hardware
//
#include <hip/hip_runtime.h>

typedef __attribute__((ext_vector_type(16))) __bf16 v16bf;
typedef __attribute__((ext_vector_type(8)))  __bf16 v8bf;
typedef __attribute__((ext_vector_type(8)))  float  v8f;
typedef __attribute__((ext_vector_type(4)))  float  v4f;

#define DIM 128

// ---------------- fp32 -> bf16 weight conversion ----------------
__global__ void cvt_bf16_kernel(const float* __restrict__ src, __bf16* __restrict__ dst, int n) {
  int i = blockIdx.x * blockDim.x + threadIdx.x;
  if (i < n) dst[i] = (__bf16)src[i];
}

// ---------------- edge scatter: agg[dst] += x_src[src]; cnt[dst] += 1 ----------------
// one wave32 per edge; each lane handles 4 contiguous floats (float4 load, 4 f32 atomics).
// Working set (x_src + agg) is fully L2-resident (192 MB), so atomics resolve in L2.
__global__ void scatter_kernel(const float* __restrict__ xsrc, const int* __restrict__ src,
                               const int* __restrict__ dst, float* __restrict__ agg,
                               float* __restrict__ cnt, int E) {
  int t = blockIdx.x * blockDim.x + threadIdx.x;
  int e = t >> 5;
  int lane = t & 31;
  if (e >= E) return;
  int s = src[e];
  int d = dst[e];
  v4f v = *(const v4f*)(xsrc + (size_t)s * DIM + lane * 4);
  float* base = agg + (size_t)d * DIM + lane * 4;
  unsafeAtomicAdd(base + 0, v[0]);
  unsafeAtomicAdd(base + 1, v[1]);
  unsafeAtomicAdd(base + 2, v[2]);
  unsafeAtomicAdd(base + 3, v[3]);
  if (lane == 0) unsafeAtomicAdd(cnt + d, 1.0f);
}

// ---------------- fused SAGE GEMM: acc += (agg/cnt)@Wl^T + bl + xdst@Wr^T ----------------
// block = 256 threads = 8 waves; block covers 16 rows; wave w covers cols [16w,16w+16).
// Stage A (mean, x_dst) once per block in LDS as bf16 (shared across the 8 waves),
// rows padded to 136 bf16 (68 dwords) so 16B fragment reads stride 4 banks/lane ->
// all 64 banks covered, conflict-free ds_load_b128.
// A fragment (16x32 bf16): lane L (lo=L&15, hi=L>>4) holds row (row0+lo),
//   K = {kb..kb+7, kb+16..kb+23} with kb = 32*chunk + 8*hi   (ISA 05_wmma layout)
// B fragment (32x16 bf16) for X@W^T: lane L holds W row (col0+lo), same K pattern ->
//   two contiguous 16-byte bf16 loads from the preconverted weights.
__global__ __launch_bounds__(256)
void sage_gemm_kernel(const float* __restrict__ agg, const float* __restrict__ cnt,
                      const float* __restrict__ xdst,
                      const __bf16* __restrict__ wlb, const __bf16* __restrict__ wrb,
                      const float* __restrict__ bl, float* __restrict__ acc) {
  constexpr int LDST = DIM + 8;          // 136 bf16 = 272 B = 68 dwords per row
  __shared__ __bf16 smA[16 * LDST];      // mean tile, pre-scaled by 1/cnt
  __shared__ __bf16 smX[16 * LDST];      // x_dst tile

  const int t = threadIdx.x;
  const int row0 = blockIdx.x * 16;

  // cooperative stage: 16 rows x 32 float4 = 512 float4 per matrix; 2 per thread
#pragma unroll
  for (int q = t; q < 512; q += 256) {
    const int r = q >> 5;
    const int c = (q & 31) * 4;
    const float inv = 1.0f / fmaxf(cnt[row0 + r], 1.0f);
    v4f m = *(const v4f*)(agg  + (size_t)(row0 + r) * DIM + c);
    v4f x = *(const v4f*)(xdst + (size_t)(row0 + r) * DIM + c);
    __bf16* pa = smA + r * LDST + c;
    __bf16* px = smX + r * LDST + c;
#pragma unroll
    for (int j = 0; j < 4; ++j) {
      pa[j] = (__bf16)(m[j] * inv);
      px[j] = (__bf16)x[j];
    }
  }
  __syncthreads();

  const int wave = t >> 5;
  const int lane = t & 31;
  const int lo = lane & 15;
  const int hi = lane >> 4;
  const int col0 = wave * 16;

  const __bf16* wlr = wlb + (size_t)(col0 + lo) * DIM;
  const __bf16* wrr = wrb + (size_t)(col0 + lo) * DIM;
  const __bf16* la  = smA + lo * LDST;
  const __bf16* lx  = smX + lo * LDST;

  v8f d = {};
#pragma unroll
  for (int ch = 0; ch < 4; ++ch) {
    const int kb = ch * 32 + hi * 8;
    union { v16bf v; v8bf h[2]; } a, b;

    // ---- mean @ Wl^T contribution ----
    a.h[0] = *(const v8bf*)(la + kb);
    a.h[1] = *(const v8bf*)(la + kb + 16);
    b.h[0] = *(const v8bf*)(wlr + kb);
    b.h[1] = *(const v8bf*)(wlr + kb + 16);
    d = __builtin_amdgcn_wmma_f32_16x16x32_bf16(false, a.v, false, b.v, (short)0, d, false, false);

    // ---- x_dst @ Wr^T contribution ----
    a.h[0] = *(const v8bf*)(lx + kb);
    a.h[1] = *(const v8bf*)(lx + kb + 16);
    b.h[0] = *(const v8bf*)(wrr + kb);
    b.h[1] = *(const v8bf*)(wrr + kb + 16);
    d = __builtin_amdgcn_wmma_f32_16x16x32_bf16(false, a.v, false, b.v, (short)0, d, false, false);
  }

  // D layout: lane L, vgpr r -> (M = r + 8*hi, N = lo). Unique tile ownership -> plain RMW.
  const float bj = bl[col0 + lo];
#pragma unroll
  for (int r = 0; r < 8; ++r) {
    float* p = acc + (size_t)(row0 + hi * 8 + r) * DIM + col0 + lo;
    *p += d[r] + bj;
  }
}

// ---------------- x = relu(x * scale) ----------------
__global__ void finalize_kernel(float* __restrict__ x, float scale, int n) {
  int i = blockIdx.x * blockDim.x + threadIdx.x;
  if (i < n) x[i] = fmaxf(x[i] * scale, 0.0f);
}

extern "C" void kernel_launch(void* const* d_in, const int* in_sizes, int n_in,
                              void* d_out, int out_size, void* d_ws, size_t ws_size,
                              hipStream_t stream) {
  (void)in_sizes; (void)n_in; (void)out_size; (void)ws_size;

  const int    NN[3]  = {20000, 10000, 50000};                 // drug, disease, gene
  const size_t seg[3] = {0, 20000ull * DIM, 30000ull * DIM};   // offsets in concat output
  const int    stp[5] = {0, 2, 2, 1, 2};                       // src node type per edge set
  const int    dtp[5] = {2, 0, 1, 2, 2};                       // dst node type per edge set
  const int    NE[5]  = {640000, 640000, 320000, 320000, 800000};

  // workspace layout (floats): xA[80000*128] | agg[50000*128] | cnt[50176] | wlb,wrb bf16
  float*  xA   = (float*)d_ws;
  float*  aggb = xA + 80000ull * DIM;
  float*  cntb = aggb + 50000ull * DIM;
  __bf16* wlb  = (__bf16*)(cntb + 50176);
  __bf16* wrb  = wlb + 2 * 5 * DIM * DIM;

  const int NW = 2 * 5 * DIM * DIM;  // 163840 per weight tensor
  cvt_bf16_kernel<<<(NW + 255) / 256, 256, 0, stream>>>((const float*)d_in[3], wlb, NW);
  cvt_bf16_kernel<<<(NW + 255) / 256, 256, 0, stream>>>((const float*)d_in[4], wrb, NW);

  const float* bl = (const float*)d_in[5];

  for (int l = 0; l < 2; ++l) {
    const float* xin[3];
    float* accB;
    if (l == 0) {
      xin[0] = (const float*)d_in[0];
      xin[1] = (const float*)d_in[1];
      xin[2] = (const float*)d_in[2];
      accB = xA;
    } else {
      xin[0] = xA + seg[0];
      xin[1] = xA + seg[1];
      xin[2] = xA + seg[2];
      accB = (float*)d_out;
    }
    hipMemsetAsync(accB, 0, 80000ull * DIM * sizeof(float), stream);

    for (int ei = 0; ei < 5; ++ei) {
      const int nd = NN[dtp[ei]];
      hipMemsetAsync(aggb, 0, (size_t)nd * DIM * sizeof(float), stream);
      hipMemsetAsync(cntb, 0, (size_t)nd * sizeof(float), stream);

      const int* esrc = (const int*)d_in[6 + 2 * ei];
      const int* edst = (const int*)d_in[7 + 2 * ei];
      const int nthr = NE[ei] * 32;
      scatter_kernel<<<(nthr + 255) / 256, 256, 0, stream>>>(
          xin[stp[ei]], esrc, edst, aggb, cntb, NE[ei]);

      const int wi = l * 5 + ei;
      sage_gemm_kernel<<<nd / 16, 256, 0, stream>>>(
          aggb, cntb, xin[dtp[ei]],
          wlb + (size_t)wi * DIM * DIM, wrb + (size_t)wi * DIM * DIM,
          bl + (size_t)wi * DIM, accB + seg[dtp[ei]]);
    }

    // HeteroConv mean over edge types (drug:1, disease:1, gene:3) then ReLU
    finalize_kernel<<<(20000 * DIM + 255) / 256, 256, 0, stream>>>(accB + seg[0], 1.0f, 20000 * DIM);
    finalize_kernel<<<(10000 * DIM + 255) / 256, 256, 0, stream>>>(accB + seg[1], 1.0f, 10000 * DIM);
    finalize_kernel<<<(50000 * DIM + 255) / 256, 256, 0, stream>>>(accB + seg[2], 1.0f / 3.0f, 50000 * DIM);
  }
}
